// RNNEncoder_46978352284231
// MI455X (gfx1250) — matrix operand order repaired and checked by reading, not compiled, not run on hardware
//
#include <hip/hip_runtime.h>

// RNN GRU encoder for gfx1250 (MI455X), wave32 + v_wmma_f32_16x16x32_bf16.
// T=128, B=2048, H=256, V=64, L2OUT=100.
//
// Pipeline:
//  k_compute_P : P[v][3H] = emb[v] @ w_ih0^T + b_ih0, stored bf16 (96 KB)
//  k_repack    : f32 [768,256] weight -> bf16 fragment-major pack matching
//                the WMMA B-matrix (32x16, K x N) per-lane layout.
//  k_gru       : 64 WGs x 256 thr; each WG owns 32 batch rows, runs all 128
//                timesteps of both GRU layers. LDS-resident: h1/h2 state
//                (double-buffered bf16, bank-conflict-padded), P table (bf16),
//                all tokens for the WG, biases. Weights stream from L2 as
//                pre-packed b128 fragments. Biases are folded into WMMA
//                accumulator init. Fuses the t==lens[b] gather into out_h.
//  k_fc        : out_h @ w1^T + b1 -> LeakyReLU -> @ w2^T + b2 -> enc.

typedef __attribute__((ext_vector_type(16))) __bf16 v16bf;
typedef __attribute__((ext_vector_type(8)))  float  v8f;

#define Tsteps 128
#define Bsz    2048
#define Hdim   256
#define HPAD   264     // padded LDS row stride (halfs): 528B = 132 dwords ->
                       // lane i of a b128 A-frag load starts at bank (4*i)%64
#define G3H    768
#define Vvoc   64
#define L2OUT  100
#define BTILE  32      // batch rows per workgroup
#define NWAVES 8

union FragU {
    uint4 u[2];
    v16bf v;
};

static __device__ __forceinline__ float bf2f(unsigned short x) {
    unsigned u = ((unsigned)x) << 16;
    return __builtin_bit_cast(float, u);
}
static __device__ __forceinline__ unsigned short f2bf(float f) {
    unsigned u = __builtin_bit_cast(unsigned, f);
    u = (u + 0x7FFFu + ((u >> 16) & 1u)) >> 16;   // round-to-nearest-even
    return (unsigned short)u;
}
static __device__ __forceinline__ float sigmoidf(float x) {
    return 1.0f / (1.0f + __expf(-x));
}
static __device__ __forceinline__ v8f splat8(float x) {
    v8f v = {x, x, x, x, x, x, x, x};
    return v;
}

static __device__ __forceinline__ v8f wmma_bf16(v16bf a, v16bf b, v8f c) {
    return __builtin_amdgcn_wmma_f32_16x16x32_bf16(
        /*neg_a=*/false, a, /*neg_b=*/false, b,
        /*c_mod=*/(short)0, c, /*reuse_a=*/false, /*reuse_b=*/false);
}

// A fragment: 16x32 bf16 (M x K) from LDS, rows are batch rows (stride HPAD).
// ISA layout: lane<16 holds M=lane, K in {k..k+7} u {k+16..k+23};
//             lane>=16 holds M=lane-16, K in {k+8..k+15} u {k+24..k+31}.
static __device__ __forceinline__ v16bf load_A_lds(const unsigned short* base,
                                                   int k, int lane) {
    int row = lane & 15;
    int kb  = k * 32 + ((lane & 16) ? 8 : 0);
    const uint4* p = reinterpret_cast<const uint4*>(base + row * HPAD + kb);
    FragU f;
    f.u[0] = p[0];   // K = kb .. kb+7
    f.u[1] = p[2];   // K = kb+16 .. kb+23
    return f.v;
}

// B fragment: 32x16 bf16 (K x N); pre-packed fragment-major: each lane's
// 16 halves are contiguous (two global b128 loads, base + imm offsets).
static __device__ __forceinline__ v16bf load_B_pk(const unsigned short* wbase,
                                                  int nt, int k, int lane) {
    const uint4* p = reinterpret_cast<const uint4*>(
        wbase + ((nt * 8 + k) * 512 + lane * 16));
    FragU f;
    f.u[0] = p[0];
    f.u[1] = p[1];
    return f.v;
}

// ---------- P = emb @ w_ih0^T + b_ih0  ([64, 768], stored bf16) ----------
__global__ __launch_bounds__(256) void k_compute_P(
    const float* __restrict__ emb, const float* __restrict__ w_ih0,
    const float* __restrict__ b_ih0, unsigned short* __restrict__ Pb) {
    int idx = blockIdx.x * 256 + threadIdx.x;      // 0 .. 64*768-1
    if (idx >= Vvoc * G3H) return;
    int v = idx / G3H;
    int c = idx - v * G3H;
    float acc = b_ih0[c];
    const float* e = emb + v * Hdim;
    const float* w = w_ih0 + c * Hdim;
    #pragma unroll 4
    for (int kk = 0; kk < Hdim; ++kk) acc = fmaf(e[kk], w[kk], acc);
    Pb[idx] = f2bf(acc);
}

// -------- repack [768,256] f32 -> bf16 fragment-major (48 x 8 frags) ----
__global__ __launch_bounds__(256) void k_repack(
    const float* __restrict__ w, unsigned short* __restrict__ dst) {
    int idx = blockIdx.x * 256 + threadIdx.x;      // 0 .. 196607
    int i    = idx & 15;
    int lane = (idx >> 4) & 31;
    int kt   = (idx >> 9) & 7;
    int nt   = idx >> 12;                          // 0..47
    int row  = nt * 16 + (lane & 15);              // output col n (W row)
    int col  = kt * 32 + ((lane >> 4) * 8) + (i & 7) + ((i >> 3) * 16);   // K index
    dst[idx] = f2bf(w[row * Hdim + col]);
}

// ------------------------------ GRU core --------------------------------
__global__ __launch_bounds__(256) void k_gru(
    const int* __restrict__ tokens, const int* __restrict__ lens,
    const unsigned short* __restrict__ Pg,
    const unsigned short* __restrict__ Whh0p,
    const unsigned short* __restrict__ Wih1p,
    const unsigned short* __restrict__ Whh1p,
    const float* __restrict__ b_hh0, const float* __restrict__ b_ih1,
    const float* __restrict__ b_hh1, float* __restrict__ out_h) {

    __shared__ unsigned short h1s[2][BTILE * HPAD];   // 33792 B
    __shared__ unsigned short h2s[2][BTILE * HPAD];   // 33792 B
    __shared__ unsigned short Pb[Vvoc * G3H];         // 98304 B
    __shared__ float bias_s[3 * G3H];                 // [b_hh0 | b_ih1 | b_hh1]
    __shared__ int tok_s[Tsteps * BTILE];             // 16384 B
    __shared__ int len_s[BTILE];

    const int tid   = threadIdx.x;
    const int wave  = tid >> 5;
    const int lane  = tid & 31;
    const int bbase = blockIdx.x * BTILE;

    for (int i = tid; i < BTILE * HPAD; i += 256) {
        h1s[0][i] = 0;
        h2s[0][i] = 0;
    }
    for (int i = tid; i < Vvoc * G3H; i += 256) Pb[i] = Pg[i];
    for (int i = tid; i < Tsteps * BTILE; i += 256) {
        int tt = i >> 5, bb = i & 31;
        tok_s[i] = tokens[tt * Bsz + bbase + bb];
    }
    for (int i = tid; i < 3 * G3H; i += 256) {
        bias_s[i] = (i < G3H) ? b_hh0[i]
                  : (i < 2 * G3H) ? b_ih1[i - G3H] : b_hh1[i - 2 * G3H];
    }
    if (tid < BTILE) len_s[tid] = lens[bbase + tid];
    __syncthreads();

    for (int t = 0; t < Tsteps; ++t) {
        const int p = t & 1, q = p ^ 1;

        // ---------------- layer 0: gh = h1 @ Whh0^T ; gi = P[tok] --------
        for (int task = wave; task < (BTILE / 16) * 16; task += NWAVES) {
            const int rt = task >> 4;          // row tile (0..1)
            const int ct = task & 15;          // H col tile (0..15)
            const int N  = lane & 15;
            const int mo = (lane & 16) ? 8 : 0;
            const int cH = ct * 16 + N;
            // bias folded into accumulator init (bias depends only on col)
            v8f ar = splat8(bias_s[cH]);
            v8f az = splat8(bias_s[256 + cH]);
            v8f an = splat8(bias_s[512 + cH]);
            const unsigned short* habase = &h1s[p][rt * 16 * HPAD];
            #pragma unroll
            for (int k = 0; k < 8; ++k) {
                v16bf A  = load_A_lds(habase, k, lane);
                v16bf Br = load_B_pk(Whh0p, ct,      k, lane);
                v16bf Bz = load_B_pk(Whh0p, 16 + ct, k, lane);
                v16bf Bn = load_B_pk(Whh0p, 32 + ct, k, lane);
                ar = wmma_bf16(A, Br, ar);
                az = wmma_bf16(A, Bz, az);
                an = wmma_bf16(A, Bn, an);
            }
            #pragma unroll
            for (int r = 0; r < 8; ++r) {
                const int lr = rt * 16 + r + mo;
                const unsigned short* Pr = Pb + tok_s[t * BTILE + lr] * G3H + cH;
                float rg = sigmoidf(bf2f(Pr[0])   + ar[r]);
                float zg = sigmoidf(bf2f(Pr[256]) + az[r]);
                float ng = tanhf(bf2f(Pr[512]) + rg * an[r]);
                float ho = bf2f(h1s[p][lr * HPAD + cH]);
                float hn = (1.0f - zg) * ng + zg * ho;
                h1s[q][lr * HPAD + cH] = f2bf(hn);
            }
        }
        __syncthreads();

        // -------- layer 1: gi = h1' @ Wih1^T ; gh = h2 @ Whh1^T ----------
        for (int task = wave; task < (BTILE / 16) * 16; task += NWAVES) {
            const int rt = task >> 4;
            const int ct = task & 15;
            const int N  = lane & 15;
            const int mo = (lane & 16) ? 8 : 0;
            const int cH = ct * 16 + N;
            v8f arz = splat8(bias_s[G3H + cH]       + bias_s[2 * G3H + cH]);
            v8f azz = splat8(bias_s[G3H + 256 + cH] + bias_s[2 * G3H + 256 + cH]);
            v8f ani = splat8(bias_s[G3H + 512 + cH]);
            v8f anh = splat8(bias_s[2 * G3H + 512 + cH]);
            const unsigned short* h1b = &h1s[q][rt * 16 * HPAD];
            const unsigned short* h2b = &h2s[p][rt * 16 * HPAD];
            #pragma unroll
            for (int k = 0; k < 8; ++k) {
                v16bf A1 = load_A_lds(h1b, k, lane);
                v16bf A2 = load_A_lds(h2b, k, lane);
                arz = wmma_bf16(A1, load_B_pk(Wih1p, ct,      k, lane), arz);
                arz = wmma_bf16(A2, load_B_pk(Whh1p, ct,      k, lane), arz);
                azz = wmma_bf16(A1, load_B_pk(Wih1p, 16 + ct, k, lane), azz);
                azz = wmma_bf16(A2, load_B_pk(Whh1p, 16 + ct, k, lane), azz);
                ani = wmma_bf16(A1, load_B_pk(Wih1p, 32 + ct, k, lane), ani);
                anh = wmma_bf16(A2, load_B_pk(Whh1p, 32 + ct, k, lane), anh);
            }
            #pragma unroll
            for (int r = 0; r < 8; ++r) {
                const int lr = rt * 16 + r + mo;
                float rg = sigmoidf(arz[r]);
                float zg = sigmoidf(azz[r]);
                float ng = tanhf(ani[r] + rg * anh[r]);
                float ho = bf2f(h2s[p][lr * HPAD + cH]);
                float hn = (1.0f - zg) * ng + zg * ho;
                h2s[q][lr * HPAD + cH] = f2bf(hn);
                if (len_s[lr] == t) out_h[(bbase + lr) * Hdim + cH] = hn;
            }
        }
        __syncthreads();
    }
}

// ---------------- FC head: LeakyReLU(x@w1^T+b1)@w2^T + b2 ----------------
__global__ __launch_bounds__(256) void k_fc(
    const float* __restrict__ out_h, const float* __restrict__ w1,
    const float* __restrict__ b1, const float* __restrict__ w2,
    const float* __restrict__ b2, float* __restrict__ enc) {
    __shared__ float xs[Hdim];
    __shared__ float z1[Hdim];
    const int b = blockIdx.x;
    const int j = threadIdx.x;
    xs[j] = out_h[b * Hdim + j];
    __syncthreads();
    float acc = b1[j];
    const float* wr = w1 + j * Hdim;
    #pragma unroll 4
    for (int k = 0; k < Hdim; ++k) acc = fmaf(xs[k], wr[k], acc);
    z1[j] = (acc >= 0.0f) ? acc : 0.01f * acc;
    __syncthreads();
    if (j < L2OUT) {
        float a = b2[j];
        const float* w2r = w2 + j * Hdim;
        #pragma unroll 4
        for (int k = 0; k < Hdim; ++k) a = fmaf(z1[k], w2r[k], a);
        enc[b * L2OUT + j] = a;
    }
}

extern "C" void kernel_launch(void* const* d_in, const int* in_sizes, int n_in,
                              void* d_out, int out_size, void* d_ws, size_t ws_size,
                              hipStream_t stream) {
    const int*   tokens = (const int*)  d_in[0];
    const int*   lens   = (const int*)  d_in[1];
    const float* emb    = (const float*)d_in[2];
    const float* w_ih0  = (const float*)d_in[3];
    const float* w_hh0  = (const float*)d_in[4];
    const float* b_ih0  = (const float*)d_in[5];
    const float* b_hh0  = (const float*)d_in[6];
    const float* w_ih1  = (const float*)d_in[7];
    const float* w_hh1  = (const float*)d_in[8];
    const float* b_ih1  = (const float*)d_in[9];
    const float* b_hh1  = (const float*)d_in[10];
    const float* w1     = (const float*)d_in[11];
    const float* b1     = (const float*)d_in[12];
    const float* w2     = (const float*)d_in[13];
    const float* b2     = (const float*)d_in[14];
    float* enc = (float*)d_out;

    char* ws = (char*)d_ws;
    unsigned short* Pb    = (unsigned short*)ws;                      // 98304 B
    unsigned short* Wpk   = (unsigned short*)(ws + 98304);            // 3*393216 B
    float*          out_h = (float*)(ws + 98304 + 3 * 393216);        // 2 MB

    const int WELEM = G3H * Hdim;   // 196608 ushorts per packed matrix

    k_compute_P<<<(Vvoc * G3H) / 256, 256, 0, stream>>>(emb, w_ih0, b_ih0, Pb);
    k_repack<<<WELEM / 256, 256, 0, stream>>>(w_hh0, Wpk);
    k_repack<<<WELEM / 256, 256, 0, stream>>>(w_ih1, Wpk + WELEM);
    k_repack<<<WELEM / 256, 256, 0, stream>>>(w_hh1, Wpk + 2 * WELEM);
    k_gru<<<Bsz / BTILE, 256, 0, stream>>>(tokens, lens, Pb,
                                           Wpk, Wpk + WELEM, Wpk + 2 * WELEM,
                                           b_hh0, b_ih1, b_hh1, out_h);
    k_fc<<<Bsz, 256, 0, stream>>>(out_h, w1, b1, w2, b2, enc);
}
